// ReweightSample_24962349924818
// MI455X (gfx1250) — matrix operand order repaired and checked by reading, not compiled, not run on hardware
//
#include <hip/hip_runtime.h>
#include <hip/hip_bf16.h>

typedef __attribute__((ext_vector_type(16))) _Float16 v16h;
typedef __attribute__((ext_vector_type(8)))  _Float16 v8h;
typedef __attribute__((ext_vector_type(8)))  float    v8f;

#define NN   4096
#define DD   256
#define FEPS 1e-7f
#define NEG_DIAG -10.0f

// ---------------------------------------------------------------------------
// Pass 0: L2-normalize each row of feat_k / feat_g, emit fp16 rows into ws.
// One 256-thread block per row (blockDim.x == DD). blockIdx.x in [0, 2*NN).
// ---------------------------------------------------------------------------
__global__ void normalize_rows(const float* __restrict__ fk,
                               const float* __restrict__ fg,
                               _Float16* __restrict__ fkh,
                               _Float16* __restrict__ fgh) {
  int b   = blockIdx.x;
  int row = b & (NN - 1);
  const float* src = (b < NN) ? fk  : fg;
  _Float16*    dst = (b < NN) ? fkh : fgh;

  int t = threadIdx.x;                       // 0..255 == DD
  float x  = src[(size_t)row * DD + t];
  float sq = x * x;
  // wave32 reduction
  for (int off = 16; off > 0; off >>= 1)
    sq += __shfl_down(sq, off, 32);

  __shared__ float partial[8];
  __shared__ float inv_s;
  int wave = t >> 5, lane = t & 31;
  if (lane == 0) partial[wave] = sq;
  __syncthreads();
  if (t == 0) {
    float s = 0.f;
#pragma unroll
    for (int w = 0; w < 8; ++w) s += partial[w];
    inv_s = 1.0f / (sqrtf(s) + FEPS);
  }
  __syncthreads();
  dst[(size_t)row * DD + t] = (_Float16)(x * inv_s);
}

// ---------------------------------------------------------------------------
// Pass 0b: reset per-row max accumulators (all candidates are >= 0, so we
// keep them as raw uint bits and use integer atomicMax).
// ---------------------------------------------------------------------------
__global__ void init_rowmax(unsigned int* __restrict__ rowmax) {
  int i = blockIdx.x * blockDim.x + threadIdx.x;
  if (i < NN) rowmax[i] = 0u;
}

// ---------------------------------------------------------------------------
// Pass 1: per-wave 16x16 tile of S = (fk fk^T + 1 + eps)/(fg fg^T + 1 + eps),
// diagonal = -10, via v_wmma_f32_16x16x32_f16 (8 K-steps x 2 Gram matrices).
// Block = 256 threads = 8 waves; wave w handles column-tile blockIdx.x*8 + w.
// Grid = (NN/16/8, NN/16) = (32, 256).
//
// Fragment layouts per CDNA5 ISA 7.12.2 (wave32):
//  A 16x32 f16 : lane l holds row M=l%16; halves 0..7 = K {0..7}+8*hi,
//                halves 8..15 = K {16..23}+8*hi          (hi = l>=16)
//  B 32x16 f16 : lane l holds col N=l%16; same K order as A
//  C/D 16x16   : vgpr r, lane l -> (M = r + 8*hi, N = l%16)
// ---------------------------------------------------------------------------
__global__ void simratio_wmma(const _Float16* __restrict__ fkh,
                              const _Float16* __restrict__ fgh,
                              float* __restrict__ S,
                              unsigned int* __restrict__ rowmax) {
  int lane = threadIdx.x & 31;
  int wave = threadIdx.x >> 5;
  int i0 = blockIdx.y * 16;
  int j0 = (blockIdx.x * 8 + wave) * 16;
  int lrow = lane & 15;
  int hi   = lane >> 4;          // 0 or 1

  const _Float16* ak = fkh + (size_t)(i0 + lrow) * DD;
  const _Float16* bk = fkh + (size_t)(j0 + lrow) * DD;
  const _Float16* ag = fgh + (size_t)(i0 + lrow) * DD;
  const _Float16* bg = fgh + (size_t)(j0 + lrow) * DD;

  const int aoff = hi * 8;       // A: +8 halves for upper half-wave
  const int boff = hi * 8;       // B: +8 halves for upper half-wave

  v8f ck = {};
  v8f cg = {};
#pragma unroll
  for (int k = 0; k < DD; k += 32) {
    v8h alo = *(const v8h*)(ak + k + aoff);
    v8h ahi = *(const v8h*)(ak + k + 16 + aoff);
    v16h aK = __builtin_shufflevector(alo, ahi,
              0,1,2,3,4,5,6,7,8,9,10,11,12,13,14,15);
    v16h bK = __builtin_shufflevector(*(const v8h*)(bk + k + boff), *(const v8h*)(bk + k + 16 + boff),
              0,1,2,3,4,5,6,7,8,9,10,11,12,13,14,15);
    ck = __builtin_amdgcn_wmma_f32_16x16x32_f16(
        false, aK, false, bK, (short)0, ck, false, false);

    v8h glo = *(const v8h*)(ag + k + aoff);
    v8h ghi = *(const v8h*)(ag + k + 16 + aoff);
    v16h aG = __builtin_shufflevector(glo, ghi,
              0,1,2,3,4,5,6,7,8,9,10,11,12,13,14,15);
    v16h bG = __builtin_shufflevector(*(const v8h*)(bg + k + boff), *(const v8h*)(bg + k + 16 + boff),
              0,1,2,3,4,5,6,7,8,9,10,11,12,13,14,15);
    cg = __builtin_amdgcn_wmma_f32_16x16x32_f16(
        false, aG, false, bG, (short)0, cg, false, false);
  }

  // Epilogue: ratio, diagonal, store, per-row running max.
#pragma unroll
  for (int r = 0; r < 8; ++r) {
    int grow = i0 + r + hi * 8;
    int gcol = j0 + lrow;
    float s = (ck[r] + 1.0f + FEPS) / (cg[r] + 1.0f + FEPS);
    if (grow == gcol) s = NEG_DIAG;
    S[(size_t)grow * NN + gcol] = s;

    float m = fmaxf(s, 0.0f);    // off-diag entries are positive; -10 ignored
    // reduce across the 16 lanes sharing this output row
    for (int off = 8; off > 0; off >>= 1)
      m = fmaxf(m, __shfl_xor(m, off, 32));
    if (lrow == 0)
      atomicMax(rowmax + grow, __float_as_uint(m));
  }
}

// ---------------------------------------------------------------------------
// Pass 2: in-place out[i][j] = exp(s_ij - rowmax_i). float4 vectorized.
// ---------------------------------------------------------------------------
__global__ void finalize_exp(float* __restrict__ S,
                             const unsigned int* __restrict__ rowmax) {
  size_t idx = ((size_t)blockIdx.x * blockDim.x + threadIdx.x) * 4;
  int row = (int)(idx >> 12);                // idx / NN
  float m = __uint_as_float(rowmax[row]);
  float4 v = *(float4*)(S + idx);
  v.x = __expf(v.x - m);
  v.y = __expf(v.y - m);
  v.z = __expf(v.z - m);
  v.w = __expf(v.w - m);
  *(float4*)(S + idx) = v;
}

extern "C" void kernel_launch(void* const* d_in, const int* in_sizes, int n_in,
                              void* d_out, int out_size, void* d_ws, size_t ws_size,
                              hipStream_t stream) {
  // setup_inputs order: feat_q (unused for output), feat_k, feat_g
  const float* feat_k = (const float*)d_in[1];
  const float* feat_g = (const float*)d_in[2];
  float* out = (float*)d_out;

  char* ws = (char*)d_ws;
  _Float16* fkh = (_Float16*)ws;                                     // 2 MB
  _Float16* fgh = (_Float16*)(ws + (size_t)NN * DD * sizeof(_Float16)); // 2 MB
  unsigned int* rowmax =
      (unsigned int*)(ws + 2 * (size_t)NN * DD * sizeof(_Float16));  // 16 KB

  normalize_rows<<<2 * NN, DD, 0, stream>>>(feat_k, feat_g, fkh, fgh);
  init_rowmax<<<NN / 256, 256, 0, stream>>>(rowmax);

  dim3 grid(NN / 16 / 8, NN / 16);   // (32, 256), 8 waves per block
  simratio_wmma<<<grid, 256, 0, stream>>>(fkh, fgh, out, rowmax);

  size_t elems = (size_t)NN * NN;    // 16,777,216
  finalize_exp<<<(unsigned)(elems / 4 / 256), 256, 0, stream>>>(out, rowmax);
}
